// SelfAttentionWithString3D_25005299597938
// MI455X (gfx1250) — matrix operand order repaired and checked by reading, not compiled, not run on hardware
//
#include <hip/hip_runtime.h>

// ---------------------------------------------------------------------------
// Types for CDNA5 WMMA bf16 (wave32): D(f32 16x16) = A(16x32 bf16) x B(32x16 bf16) + C
// ---------------------------------------------------------------------------
typedef __attribute__((ext_vector_type(16))) __bf16 v16bf;
typedef __attribute__((ext_vector_type(8)))  __bf16 v8bf;
typedef __attribute__((ext_vector_type(8)))  float  v8f;

#define BATCH 64
#define SEQ   197
#define SP    208      // Q/K padded sequence (13 * 16)
#define VTS   224      // V^T padded key stride (7 * 32)
#define EMB   768
#define NH    8
#define HDIM  96
#define MTOT  (BATCH*SEQ)   // 12608 = 197 * 64 rows

__device__ __forceinline__ __bf16 f2bf(float f) { return (__bf16)f; }

__device__ __forceinline__ v8f vzero8() {
  v8f r;
#pragma unroll
  for (int i = 0; i < 8; i++) r[i] = 0.f;
  return r;
}

__device__ __forceinline__ v16bf ld16(const __bf16* p0, const __bf16* p1) {
  v8bf a = *(const v8bf*)p0;
  v8bf b = *(const v8bf*)p1;
  v16bf r;
#pragma unroll
  for (int i = 0; i < 8; i++) { r[i] = a[i]; r[8 + i] = b[i]; }
  return r;
}

// A-matrix 16x32 bf16 tile: lane row = lane&15; K = kb..kb+7 and 16+kb..16+kb+7, kb=(lane>>4)*8
__device__ __forceinline__ v16bf loadA(const __bf16* row, int lane) {
  int kb = (lane >> 4) * 8;
  return ld16(row + kb, row + 16 + kb);
}
// B-matrix 32x16 bf16 tile: lane col = lane&15; K = kb..kb+7 and 16+kb..16+kb+7, kb=(lane>>4)*8
__device__ __forceinline__ v16bf loadB(const __bf16* row, int lane) {
  int kb = (lane >> 4) * 8;
  return ld16(row + kb, row + 16 + kb);
}

__device__ __forceinline__ v8f wmma_bf16(v16bf a, v16bf b, v8f c) {
  return __builtin_amdgcn_wmma_f32_16x16x32_bf16(false, a, false, b, (short)0, c, false, false);
}

// ---------------------------------------------------------------------------
// Cayley: P = (I+A)^-1 (I-A), A = (S - S^T)/2, 32x32, Gauss-Jordan in LDS.
// blockIdx.x in {0,1,2} selects Sx/Sy/Sz; 1024 threads = (row,col).
// ---------------------------------------------------------------------------
__global__ __launch_bounds__(1024) void cayley_kernel(const float* Sx, const float* Sy,
                                                      const float* Sz, float* P3) {
  __shared__ float M1[32][33];
  __shared__ float M2[32][33];
  const float* S = (blockIdx.x == 0) ? Sx : ((blockIdx.x == 1) ? Sy : Sz);
  float* P = P3 + blockIdx.x * 1024;
  int r = threadIdx.x >> 5, c = threadIdx.x & 31;
  float a = 0.5f * (S[r * 32 + c] - S[c * 32 + r]);
  float id = (r == c) ? 1.f : 0.f;
  M1[r][c] = id + a;
  M2[r][c] = id - a;
  __syncthreads();
  for (int k = 0; k < 32; k++) {
    if (r == k) {
      float inv = 1.f / M1[k][k];   // lockstep wave read-before-write
      M1[k][c] *= inv;
      M2[k][c] *= inv;
    }
    __syncthreads();
    float f = M1[r][k];
    __syncthreads();
    if (r != k) {
      M1[r][c] -= f * M1[k][c];
      M2[r][c] -= f * M2[k][c];
    }
    __syncthreads();
  }
  P[r * 32 + c] = M2[r][c];
}

// ---------------------------------------------------------------------------
// Fold P into Wq/Wk per 32-col axis block, transpose all weights, convert bf16.
// WT[n][k]: for Wq/Wk: sum_j W[k][c0+j] * P_axis[jj][j]  (n = c0+jj)
// ---------------------------------------------------------------------------
__global__ void prep_w_kernel(const float* Wq, const float* Wk, const float* Wv, const float* Wo,
                              const float* P3, __bf16* WqT, __bf16* WkT, __bf16* WvT, __bf16* WoT) {
  int idx = blockIdx.x * blockDim.x + threadIdx.x;
  if (idx >= EMB * EMB) return;
  int n = idx / EMB, k = idx % EMB;
  int c0 = (n >> 5) << 5;
  int jj = n & 31;
  int axis = (n % HDIM) / 32;
  const float* P = P3 + axis * 1024 + jj * 32;
  const float* wqr = Wq + (size_t)k * EMB + c0;
  const float* wkr = Wk + (size_t)k * EMB + c0;
  float sq = 0.f, sk = 0.f;
#pragma unroll 8
  for (int j = 0; j < 32; j++) {
    float p = P[j];
    sq += wqr[j] * p;
    sk += wkr[j] * p;
  }
  WqT[(size_t)n * EMB + k] = f2bf(sq);
  WkT[(size_t)n * EMB + k] = f2bf(sk);
  WvT[(size_t)n * EMB + k] = f2bf(Wv[(size_t)k * EMB + n]);
  WoT[(size_t)n * EMB + k] = f2bf(Wo[(size_t)k * EMB + n]);
}

__global__ void convert_x_kernel(const float* x, __bf16* xb) {
  size_t n = (size_t)MTOT * EMB;
  for (size_t i = (size_t)blockIdx.x * blockDim.x + threadIdx.x; i < n;
       i += (size_t)gridDim.x * blockDim.x)
    xb[i] = f2bf(x[i]);
}

// ---------------------------------------------------------------------------
// Depth map -> per-patch batch-mean z coordinate (196 patches + cls=0)
// ---------------------------------------------------------------------------
__global__ __launch_bounds__(256) void depth_z_kernel(const float* depth, const float* dzw,
                                                      const float* dzb, float* zc) {
  __shared__ float red[256];
  int p = blockIdx.x;
  int py = p / 14, px = p % 14;
  int t = threadIdx.x;
  int i = t >> 4, j = t & 15;
  float s = 0.f;
#pragma unroll 4
  for (int b = 0; b < BATCH; b++)
    s += depth[(size_t)b * 224 * 224 + (py * 16 + i) * 224 + (px * 16 + j)];
  red[t] = s;
  __syncthreads();
  for (int st = 128; st > 0; st >>= 1) {
    if (t < st) red[t] += red[t + st];
    __syncthreads();
  }
  if (t == 0) zc[1 + p] = dzw[0] * (red[0] * (1.f / (64.f * 256.f))) + dzb[0];
  if (blockIdx.x == 0 && t == 0) zc[0] = 0.f;
}

// cos/sin tables (SEQ x HDIM): axis = d/32, repeated-freq index = (d%32)>>1
__global__ void trig_kernel(const float* zc, float* cosT, float* sinT) {
  int idx = blockIdx.x * blockDim.x + threadIdx.x;
  if (idx >= SEQ * HDIM) return;
  int s = idx / HDIM, d = idx % HDIM;
  int axis = d / 32, jj = d & 31, fi = jj >> 1;
  float freq = __powf(10000.f, -(float)fi * (1.f / 16.f));
  float coord;
  if (axis == 0)      coord = (s == 0) ? 0.f : (float)((s - 1) % 14 + 1);
  else if (axis == 1) coord = (s == 0) ? 0.f : (float)((s - 1) / 14 + 1);
  else                coord = zc[s];
  float ang = coord * freq;
  cosT[idx] = __cosf(ang);
  sinT[idx] = __sinf(ang);
}

// ---------------------------------------------------------------------------
// WMMA GEMM: C[12608 x 768] = A(bf16, row-major) x BT(bf16, transposed weights)
// 4 waves/block; wave computes 16 rows x 64 cols (4 accumulators).
// MODE 0: store V^T bf16 [(b,h), d, s(pad 224)]
// MODE 1: RoPE epilogue, store Q/K bf16 [(b,h), s(pad 208), d]
// MODE 2: +bias, store f32 row-major (final output)
// ---------------------------------------------------------------------------
template <int MODE>
__global__ __launch_bounds__(128) void gemm_wmma_kernel(const __bf16* __restrict__ A,
                                                        const __bf16* __restrict__ BT,
                                                        const float* __restrict__ cosT,
                                                        const float* __restrict__ sinT,
                                                        const float* __restrict__ bias,
                                                        __bf16* __restrict__ outBf,
                                                        float* __restrict__ outF) {
  int lane = threadIdx.x & 31;
  int wave = threadIdx.x >> 5;
  int m0 = blockIdx.y * 64 + wave * 16;
  int n0 = blockIdx.x * 64;
  int am = lane & 15;

  v8f acc[4];
#pragma unroll
  for (int t = 0; t < 4; t++) acc[t] = vzero8();

  const __bf16* arow = A + (size_t)(m0 + am) * EMB;
  const __bf16* brow0 = BT + (size_t)(n0 + am) * EMB;

  for (int kk = 0; kk < EMB; kk += 32) {
    v16bf a = loadA(arow + kk, lane);
#pragma unroll
    for (int t = 0; t < 4; t++) {
      v16bf b = loadB(brow0 + (size_t)(t * 16) * EMB + kk, lane);
      acc[t] = wmma_bf16(a, b, acc[t]);
    }
  }

  int mrow0 = m0 + ((lane >> 4) * 8);
  if (MODE == 2) {
#pragma unroll
    for (int t = 0; t < 4; t++) {
      int col = n0 + t * 16 + am;
      float bv = bias[col];
#pragma unroll
      for (int i = 0; i < 8; i++)
        outF[(size_t)(mrow0 + i) * EMB + col] = acc[t][i] + bv;
    }
  } else if (MODE == 0) {
#pragma unroll
    for (int t = 0; t < 4; t++) {
      int col = n0 + t * 16 + am;
      int h = col / HDIM, dd = col % HDIM;
#pragma unroll
      for (int i = 0; i < 8; i++) {
        int m = mrow0 + i;
        int s = m % SEQ, b = m / SEQ;
        outBf[((size_t)(b * NH + h) * HDIM + dd) * VTS + s] = f2bf(acc[t][i]);
      }
    }
  } else {  // MODE 1: per-token RoPE; pair (d, d+16) lives in adjacent 16-col tile, same lane
#pragma unroll
    for (int tp = 0; tp < 2; tp++) {
      int col_e = n0 + tp * 32 + am;          // (col_e % 32) < 16
      int h = col_e / HDIM, dd = col_e % HDIM;
#pragma unroll
      for (int i = 0; i < 8; i++) {
        int m = mrow0 + i;
        int s = m % SEQ, b = m / SEQ;
        float ce = cosT[s * HDIM + dd],      se = sinT[s * HDIM + dd];
        float co = cosT[s * HDIM + dd + 16], so = sinT[s * HDIM + dd + 16];
        float ve = acc[tp * 2][i], vo = acc[tp * 2 + 1][i];
        size_t base = ((size_t)(b * NH + h) * SP + s) * HDIM;
        outBf[base + dd]      = f2bf(ve * ce - vo * se);
        outBf[base + dd + 16] = f2bf(vo * co + ve * so);
      }
    }
  }
}

// ---------------------------------------------------------------------------
// Attention: grid (B*H, 4) x 128 threads; each wave owns one 16-query tile.
// scores = Q K^T / sqrt(96) (13x3 WMMA), register softmax (shfl over 16 lanes),
// probs -> LDS bf16 -> A-layout reload -> P x V^T (7x6 WMMA).
// ---------------------------------------------------------------------------
__global__ __launch_bounds__(128) void attn_kernel(const __bf16* __restrict__ Qb,
                                                   const __bf16* __restrict__ Kb,
                                                   const __bf16* __restrict__ Vt,
                                                   __bf16* __restrict__ Ob) {
  __shared__ __bf16 probs[4][16][VTS];
  int bh = blockIdx.x;
  int b = bh >> 3, h = bh & 7;
  int wave = threadIdx.x >> 5, lane = threadIdx.x & 31;
  int qt = blockIdx.y * 4 + wave;
  if (qt >= 13) return;  // 13 query tiles of 16 (no __syncthreads below)
  int am = lane & 15;

  const __bf16* Qrow = Qb + ((size_t)bh * SP + qt * 16 + am) * HDIM;
  v16bf qa[3];
#pragma unroll
  for (int c = 0; c < 3; c++) qa[c] = loadA(Qrow + c * 32, lane);

  v8f sc[13];
#pragma unroll
  for (int kt = 0; kt < 13; kt++) {
    v8f a = vzero8();
    const __bf16* Krow = Kb + ((size_t)bh * SP + kt * 16 + am) * HDIM;
#pragma unroll
    for (int c = 0; c < 3; c++) {
      v16bf kb16 = loadB(Krow + c * 32, lane);   // B[d][key]: contiguous d from K row
      a = wmma_bf16(qa[c], kb16, a);
    }
    sc[kt] = a;
  }

  const float scale = 0.10206207262f;  // 1/sqrt(96)
#pragma unroll
  for (int kt = 0; kt < 13; kt++)
#pragma unroll
    for (int i = 0; i < 8; i++) sc[kt][i] *= scale;
  if (am >= 5) {  // keys 197..207 are padding
#pragma unroll
    for (int i = 0; i < 8; i++) sc[12][i] = -1e30f;
  }

  int row = (lane >> 4) * 8;
#pragma unroll
  for (int i = 0; i < 8; i++) {
    float mx = -1e30f;
#pragma unroll
    for (int kt = 0; kt < 13; kt++) mx = fmaxf(mx, sc[kt][i]);
#pragma unroll
    for (int d = 1; d < 16; d <<= 1) mx = fmaxf(mx, __shfl_xor(mx, d, 32));
    float sum = 0.f;
#pragma unroll
    for (int kt = 0; kt < 13; kt++) {
      float p = __expf(sc[kt][i] - mx);
      sc[kt][i] = p;
      sum += p;
    }
#pragma unroll
    for (int d = 1; d < 16; d <<= 1) sum += __shfl_xor(sum, d, 32);
    float inv = 1.f / sum;
#pragma unroll
    for (int kt = 0; kt < 13; kt++)
      probs[wave][row + i][kt * 16 + am] = f2bf(sc[kt][i] * inv);
    probs[wave][row + i][208 + am] = f2bf(0.f);  // zero K-pad 208..223
  }

  asm volatile("s_wait_dscnt 0" ::: "memory");   // LDS RAW within wave
  __builtin_amdgcn_wave_barrier();

  v8f oacc[6];
#pragma unroll
  for (int t = 0; t < 6; t++) oacc[t] = vzero8();
  for (int c = 0; c < 7; c++) {  // K-dim 224 = 7 x 32 keys
    v16bf pa = loadA(&probs[wave][am][c * 32], lane);
    const __bf16* vbase = Vt + ((size_t)bh * HDIM + am) * VTS + c * 32;
#pragma unroll
    for (int t = 0; t < 6; t++) {
      v16bf vb = loadB(vbase + (size_t)(t * 16) * VTS, lane);  // B[key][d]: contiguous keys from V^T row
      oacc[t] = wmma_bf16(pa, vb, oacc[t]);
    }
  }

#pragma unroll
  for (int t = 0; t < 6; t++) {
    int e = h * HDIM + t * 16 + am;
#pragma unroll
    for (int i = 0; i < 8; i++) {
      int s = qt * 16 + row + i;
      if (s < SEQ) Ob[((size_t)b * SEQ + s) * EMB + e] = f2bf(oacc[t][i]);
    }
  }
}

// ---------------------------------------------------------------------------
// Launcher
// ---------------------------------------------------------------------------
extern "C" void kernel_launch(void* const* d_in, const int* in_sizes, int n_in,
                              void* d_out, int out_size, void* d_ws, size_t ws_size,
                              hipStream_t stream) {
  const float* x     = (const float*)d_in[0];
  const float* depth = (const float*)d_in[1];
  const float* Wq    = (const float*)d_in[2];
  const float* Wk    = (const float*)d_in[3];
  const float* Wv    = (const float*)d_in[4];
  const float* Wo    = (const float*)d_in[5];
  const float* bo    = (const float*)d_in[6];
  const float* Sx    = (const float*)d_in[7];
  const float* Sy    = (const float*)d_in[8];
  const float* Sz    = (const float*)d_in[9];
  const float* dzw   = (const float*)d_in[10];
  const float* dzb   = (const float*)d_in[11];
  float* out = (float*)d_out;

  char* wsp = (char*)d_ws;
  auto take = [&](size_t bytes) -> void* {
    void* p = (void*)wsp;
    wsp += (bytes + 255) & ~(size_t)255;
    return p;
  };
  float*  P3   = (float*)take(3 * 32 * 32 * sizeof(float));
  float*  zc   = (float*)take(256 * sizeof(float));
  float*  cosT = (float*)take((size_t)SEQ * HDIM * sizeof(float));
  float*  sinT = (float*)take((size_t)SEQ * HDIM * sizeof(float));
  __bf16* Xbf  = (__bf16*)take((size_t)MTOT * EMB * 2);
  __bf16* WqT  = (__bf16*)take((size_t)EMB * EMB * 2);
  __bf16* WkT  = (__bf16*)take((size_t)EMB * EMB * 2);
  __bf16* WvT  = (__bf16*)take((size_t)EMB * EMB * 2);
  __bf16* WoT  = (__bf16*)take((size_t)EMB * EMB * 2);
  size_t qkBytes = (size_t)BATCH * NH * SP * HDIM * 2;
  size_t vtBytes = (size_t)BATCH * NH * HDIM * VTS * 2;
  __bf16* Qb = (__bf16*)take(qkBytes);
  __bf16* Kb = (__bf16*)take(qkBytes);
  __bf16* Vt = (__bf16*)take(vtBytes);
  __bf16* Ob = (__bf16*)take((size_t)MTOT * EMB * 2);

  // Small prep
  cayley_kernel<<<3, 1024, 0, stream>>>(Sx, Sy, Sz, P3);
  depth_z_kernel<<<196, 256, 0, stream>>>(depth, dzw, dzb, zc);
  trig_kernel<<<(SEQ * HDIM + 255) / 256, 256, 0, stream>>>(zc, cosT, sinT);
  prep_w_kernel<<<(EMB * EMB + 255) / 256, 256, 0, stream>>>(Wq, Wk, Wv, Wo, P3,
                                                             WqT, WkT, WvT, WoT);
  convert_x_kernel<<<2048, 256, 0, stream>>>(x, Xbf);

  // Zero padded Q/K rows and V^T key padding
  hipMemsetAsync(Qb, 0, qkBytes, stream);
  hipMemsetAsync(Kb, 0, qkBytes, stream);
  hipMemsetAsync(Vt, 0, vtBytes, stream);

  dim3 gg(EMB / 64, MTOT / 64);  // (12, 197)
  gemm_wmma_kernel<1><<<gg, 128, 0, stream>>>(Xbf, WqT, cosT, sinT, nullptr, Qb, nullptr);
  gemm_wmma_kernel<1><<<gg, 128, 0, stream>>>(Xbf, WkT, cosT, sinT, nullptr, Kb, nullptr);
  gemm_wmma_kernel<0><<<gg, 128, 0, stream>>>(Xbf, WvT, nullptr, nullptr, nullptr, Vt, nullptr);

  attn_kernel<<<dim3(BATCH * NH, 4), 128, 0, stream>>>(Qb, Kb, Vt, Ob);

  gemm_wmma_kernel<2><<<gg, 128, 0, stream>>>(Ob, WoT, nullptr, nullptr, bo, nullptr, out);

  (void)in_sizes; (void)n_in; (void)out_size; (void)ws_size;
}
